// DMFC2layers_16793322127723
// MI455X (gfx1250) — matrix operand order repaired and checked by reading, not compiled, not run on hardware
//
#include <hip/hip_runtime.h>

// ===========================================================================
// MI455X (gfx1250) implementation of the 2-layer DMFC point filter.
//
// Roofline: ~40-50 GFLOP total, ~hundreds of MB moved -> bandwidth/launch
// bound on a 23.3 TB/s part.  Strategy: activations live in HBM as f32,
// every dense channel-mixing matmul runs through V_WMMA_F32_16X16X32_F16
// (B staged f32->f16 in LDS, A pre-converted f16, f32 accumulate).  The
// contiguous-B fast path stages through GLOBAL_LOAD_ASYNC_TO_LDS_B32
// (ASYNCcnt + s_wait_asynccnt), decoupling global traffic from the WMMA
// stream.  Tiny KD=9 convs and the 128->1 heads use a plain VALU GEMM.
//
// Input order assumption: harness flattens setup_inputs() as a JAX pytree
// (dicts sorted alphabetically at every level):
//   corr_feats, idxs, params{cn1,cn2,dmfc1,dmfc2,lcc1,lcc2,pp_bn,pp_conv}, pos
// ===========================================================================

typedef __attribute__((ext_vector_type(16))) _Float16 v16h;
typedef __attribute__((ext_vector_type(8)))  _Float16 v8h;
typedef __attribute__((ext_vector_type(8)))  float    v8f;
typedef _Float16 half_t;

#define GSTRIDE(i, tot) \
  for (long i = (long)blockIdx.x * blockDim.x + threadIdx.x; i < (tot); \
       i += (long)gridDim.x * blockDim.x)

// ---------------------------------------------------------------------------
// f32 -> f16 convert (optional transpose), batched via blockIdx.z
// ---------------------------------------------------------------------------
__global__ __launch_bounds__(256) void k_cvt16(
    const float* __restrict__ s, long sB, half_t* __restrict__ d, long dB,
    int rows, int cols, long srcLd, long dstLd, int tr) {
  long b = blockIdx.z;
  const float* sb = s + b * sB;
  half_t* db = d + b * dB;
  long tot = (long)rows * cols;
  GSTRIDE(i, tot) {
    int r = (int)(i / cols), c = (int)(i % cols);
    float v = sb[(long)r * srcLd + c];
    if (tr) db[(long)c * dstLd + r] = (half_t)v;
    else    db[(long)r * dstLd + c] = (half_t)v;
  }
}

// ---------------------------------------------------------------------------
// WMMA GEMM:  Y[M x P] = alpha * (A_f16[M x K] * B) (+bias) (relu) (+resid)
// B is a logical K x P f32 matrix addressed as Bsrc[k*bsK + p*bsN].
// Block = 256 threads = 8 waves; wave w -> 16 rows; block tile 128 x 16.
// bsN==1 fast path: raw f32 tile staged by GLOBAL_LOAD_ASYNC_TO_LDS_B32,
// then repacked f32->f16 in LDS.  Fallback: branchless clamped scalar loads.
// ---------------------------------------------------------------------------
__global__ __launch_bounds__(256) void k_wmma_gemm(
    const half_t* __restrict__ A, long lda, long aBatch,
    const float* __restrict__ Bsrc, long bsK, long bsN, long bBatch,
    const float* __restrict__ bias,
    const float* __restrict__ resid, long resBatch,
    float* __restrict__ Y, long ldy, long yBatch,
    int M, int K, int P, float alpha, int relu) {
  __shared__ __align__(16) half_t Bs[16 * 72];   // 72-half col stride: 16B aligned, no bank conflicts
  __shared__ __align__(16) float  Bs32[16 * 64]; // raw f32 landing zone for async copies
  const int b     = blockIdx.z;
  const int pBase = blockIdx.x * 16;
  const int lane  = threadIdx.x & 31;
  const int l15   = lane & 15;
  const int hf    = lane >> 4;
  const int rowBase = (blockIdx.y * 8 + (int)(threadIdx.x >> 5)) * 16;

  const half_t* Ab = A + (long)b * aBatch;
  const float*  Bb = Bsrc + (long)b * bBatch;
  float*        Yb = Y + (long)b * yBatch;

  // LDS byte address of the async landing zone (flat LDS aperture keeps the
  // DS offset in the low 32 address bits).
  const unsigned ldsBase = (unsigned)(size_t)(&Bs32[0]);
  const bool fast = (bsN == 1);

  v8f acc = {};
  for (int k0 = 0; k0 < K; k0 += 64) {
    if (fast) {
      // ---- async global -> LDS staging of the raw f32 tile ----
#pragma unroll
      for (int j = 0; j < 4; ++j) {
        int f   = (int)threadIdx.x + j * 256;    // flat index kk*16 + col
        int col = f & 15;
        int kk  = f >> 4;
        int kg  = k0 + kk;
        int pg  = pBase + col;
        if (kg < K && pg < P) {
          const float* ga = Bb + (long)kg * bsK + pg;
          unsigned lo = ldsBase + (unsigned)(f * 4);
          asm volatile("global_load_async_to_lds_b32 %0, %1, off"
                       :: "v"(lo), "v"(ga) : "memory");
        } else {
          Bs32[f] = 0.0f;
        }
      }
      asm volatile("s_wait_asynccnt 0x0" ::: "memory");
      __syncthreads();
      // ---- repack f32 -> f16 fragment layout ----
      for (int f = threadIdx.x; f < 16 * 64; f += 256) {
        int col = f & 15;
        int kk  = f >> 4;
        Bs[col * 72 + kk] = (half_t)Bs32[f];
      }
    } else {
      // ---- branchless clamped scalar staging (strided B) ----
      for (int i = threadIdx.x; i < 16 * 64; i += 256) {
        int col = i & 15;
        int kk  = i >> 4;
        int kg  = k0 + kk;
        int pg  = pBase + col;
        bool inb = (kg < K) && (pg < P);
        long kc = (kg < K) ? kg : (K - 1);
        long pc = (pg < P) ? pg : (P - 1);
        float v = Bb[kc * bsK + pc * bsN];
        Bs[col * 72 + kk] = (half_t)(inb ? v : 0.0f);
      }
    }
    {  // prefetch next K-tile (global_prefetch_b8)
      int col = (int)threadIdx.x & 15;
      int kk  = (int)threadIdx.x >> 4;
      long kg = k0 + 64 + kk, pg = pBase + col;
      if (kg < K && pg < P) __builtin_prefetch(Bb + kg * bsK + pg * bsN, 0, 1);
    }
    __syncthreads();
#pragma unroll
    for (int s = 0; s < 2; ++s) {
      int ks = k0 + s * 32;
      if (ks >= K) break;                     // uniform across wave
      int row = rowBase + l15;
      int rc  = row < M ? row : M - 1;
      const half_t* ar = Ab + (long)rc * lda;
      int kc0 = ks + hf * 8;
      int kc1 = ks + 16 + hf * 8;
      v16h a;
      if (ks + 32 <= K) {                     // aligned vector path (lda % 8 == 0)
        v8h a0 = *(const v8h*)(ar + kc0);
        v8h a1 = *(const v8h*)(ar + kc1);
        a = __builtin_shufflevector(a0, a1, 0,1,2,3,4,5,6,7,8,9,10,11,12,13,14,15);
      } else {                                // ragged-K masked tail
        v8h a0, a1;
#pragma unroll
        for (int i = 0; i < 8; ++i) {
          int ka = kc0 + i, kb = kc1 + i;
          a0[i] = (ka < K) ? ar[ka] : (half_t)0.0f;
          a1[i] = (kb < K) ? ar[kb] : (half_t)0.0f;
        }
        a = __builtin_shufflevector(a0, a1, 0,1,2,3,4,5,6,7,8,9,10,11,12,13,14,15);
      }
      const v8h* bp = (const v8h*)(Bs + l15 * 72 + s * 32 + hf * 8);
      v8h b0 = bp[0], b1 = bp[2];
      v16h bf = __builtin_shufflevector(b0, b1, 0,1,2,3,4,5,6,7,8,9,10,11,12,13,14,15);
      acc = __builtin_amdgcn_wmma_f32_16x16x32_f16(false, a, false, bf,
                                                   (short)0, acc, false, false);
    }
    __syncthreads();
  }
  // epilogue; C-layout: VGPR r -> row = rowBase + r + hf*8, col = l15
#pragma unroll
  for (int r = 0; r < 8; ++r) {
    int row = rowBase + r + hf * 8;
    int col = pBase + l15;
    if (row < M && col < P) {
      float v = alpha * acc[r];
      if (bias)  v += bias[row];
      if (relu)  v = fmaxf(v, 0.0f);
      long off = (long)row * ldy + col;
      if (resid) v += resid[(long)b * resBatch + off];
      Yb[off] = v;
    }
  }
}

// ---------------------------------------------------------------------------
// Small VALU GEMM for KD=9 convs and 128->1 heads.
// act: 0 none, 1 relu, 2 sigmoid + clip[0.05,0.95]
// ---------------------------------------------------------------------------
__global__ __launch_bounds__(256) void k_gemm_small(
    const float* __restrict__ W, const float* __restrict__ bias,
    const float* __restrict__ X, long xB, long ldx,
    const float* __restrict__ resid, long resB,
    float* __restrict__ Y, long yB, long ldy,
    int M, int K, int P, int act) {
  long b = blockIdx.z;
  const float* Xb = X + b * xB;
  float* Yb = Y + b * yB;
  long tot = (long)M * P;
  GSTRIDE(i, tot) {
    int m = (int)(i / P), p = (int)(i % P);
    float s = bias ? bias[m] : 0.0f;
    for (int k = 0; k < K; ++k) s += W[m * K + k] * Xb[(long)k * ldx + p];
    if (act == 1) s = fmaxf(s, 0.0f);
    else if (act == 2) { s = 1.0f / (1.0f + __expf(-s)); s = fminf(fmaxf(s, 0.05f), 0.95f); }
    if (resid) s += resid[b * resB + (long)m * ldy + p];
    Yb[(long)m * ldy + p] = s;
  }
}

// ---------------------------------------------------------------------------
// Per-channel mean/var -> scale/shift (InstanceNorm / BatchNorm folded)
// ---------------------------------------------------------------------------
__global__ __launch_bounds__(256) void k_stats(
    const float* __restrict__ x, long chanStride, long batchStride,
    int nBatch, long L, const float* __restrict__ gamma,
    const float* __restrict__ beta, float eps,
    float* __restrict__ scale, float* __restrict__ shift) {
  int c = blockIdx.x;
  const float* base = x + (long)c * chanStride;
  float s = 0.0f, ss = 0.0f;
  for (int bb = 0; bb < nBatch; ++bb) {
    const float* p = base + (long)bb * batchStride;
    for (long i = threadIdx.x; i < L; i += blockDim.x) { float v = p[i]; s += v; ss += v * v; }
  }
  __shared__ float r1[256], r2[256];
  r1[threadIdx.x] = s; r2[threadIdx.x] = ss;
  __syncthreads();
  for (int st = 128; st > 0; st >>= 1) {
    if ((int)threadIdx.x < st) { r1[threadIdx.x] += r1[threadIdx.x + st]; r2[threadIdx.x] += r2[threadIdx.x + st]; }
    __syncthreads();
  }
  if (threadIdx.x == 0) {
    float cnt = (float)nBatch * (float)L;
    float mu  = r1[0] / cnt;
    float var = fmaxf(r2[0] / cnt - mu * mu, 0.0f);
    float g   = gamma ? gamma[c] : 1.0f;
    float bt  = beta  ? beta[c]  : 0.0f;
    float sc  = g * rsqrtf(var + eps);
    scale[c] = sc; shift[c] = bt - mu * sc;
  }
}

__global__ __launch_bounds__(256) void k_affine(
    const float* __restrict__ x, float* __restrict__ y,
    const float* __restrict__ scale, const float* __restrict__ shift,
    long total, long L, int modC, int relu) {
  GSTRIDE(i, total) {
    int ch = (int)((i / L) % modC);
    float v = x[i] * scale[ch] + shift[ch];
    if (relu) v = fmaxf(v, 0.0f);
    y[i] = v;
  }
}

// ---------------------------------------------------------------------------
// Row softmax (one block per row)
// ---------------------------------------------------------------------------
__global__ __launch_bounds__(256) void k_softmax(float* __restrict__ x, long ld, long L) {
  float* row = x + (long)blockIdx.x * ld;
  __shared__ float red[256];
  float m = -3.4e38f;
  for (long i = threadIdx.x; i < L; i += blockDim.x) m = fmaxf(m, row[i]);
  red[threadIdx.x] = m; __syncthreads();
  for (int st = 128; st > 0; st >>= 1) {
    if ((int)threadIdx.x < st) red[threadIdx.x] = fmaxf(red[threadIdx.x], red[threadIdx.x + st]);
    __syncthreads();
  }
  m = red[0]; __syncthreads();
  float s = 0.0f;
  for (long i = threadIdx.x; i < L; i += blockDim.x) { float e = __expf(row[i] - m); row[i] = e; s += e; }
  red[threadIdx.x] = s; __syncthreads();
  for (int st = 128; st > 0; st >>= 1) {
    if ((int)threadIdx.x < st) red[threadIdx.x] += red[threadIdx.x + st];
    __syncthreads();
  }
  float inv = 1.0f / red[0];
  for (long i = threadIdx.x; i < L; i += blockDim.x) row[i] *= inv;
}

// ---------------------------------------------------------------------------
// LCC helpers
// ---------------------------------------------------------------------------
__global__ __launch_bounds__(256) void k_gather_diff(
    const float* __restrict__ f, const int* __restrict__ idx,
    float* __restrict__ out, int Bn, int KD_, int N_, int Kn) {
  long tot = (long)Bn * KD_ * N_ * Kn;
  GSTRIDE(i, tot) {
    int k = (int)(i % Kn); long t = i / Kn;
    int n = (int)(t % N_); t /= N_;
    int d = (int)(t % KD_); long b = t / KD_;
    int j = idx[((long)b * N_ + n) * Kn + k];
    const float* fb = f + ((long)b * KD_ + d) * N_;
    out[i] = fb[n] - fb[j];
  }
}

__global__ __launch_bounds__(256) void k_perm(      // (b,c,n,k) -> (b,k,n,c)
    const float* __restrict__ s, float* __restrict__ d, int Bn, int C, int N_, int Kn) {
  long tot = (long)Bn * C * N_ * Kn;
  GSTRIDE(i, tot) {
    int c = (int)(i % C); long t = i / C;
    int n = (int)(t % N_); t /= N_;
    int k = (int)(t % Kn); long b = t / Kn;
    d[i] = s[(((long)b * C + c) * N_ + n) * Kn + k];
  }
}

__global__ __launch_bounds__(256) void k_tr2(       // (b,n,kd) -> (b,kd,n)
    const float* __restrict__ s, float* __restrict__ d, int Bn, int N_, int Kd) {
  long tot = (long)Bn * N_ * Kd;
  GSTRIDE(i, tot) {
    int dd = (int)(i % Kd); long t = i / Kd;
    int n = (int)(t % N_); long b = t / N_;
    d[((long)b * Kd + dd) * N_ + n] = s[i];
  }
}

__global__ __launch_bounds__(256) void k_tr32(      // batched (rows,cols) -> (cols,rows)
    const float* __restrict__ s, long sB, float* __restrict__ d, long dB, int rows, int cols) {
  long b = blockIdx.z;
  long tot = (long)rows * cols;
  GSTRIDE(i, tot) {
    int r = (int)(i / cols), c = (int)(i % cols);
    d[b * dB + (long)c * rows + r] = s[b * sB + i];
  }
}

__global__ __launch_bounds__(256) void k_copy(
    const float* __restrict__ s, long sB, float* __restrict__ d, long dB, long n) {
  long b = blockIdx.z;
  GSTRIDE(i, n) d[b * dB + i] = s[b * sB + i];
}

__global__ __launch_bounds__(256) void k_axpy(
    const float* __restrict__ x, const float* __restrict__ y, float* __restrict__ o, long n) {
  GSTRIDE(i, n) o[i] = x[i] + y[i];
}

// ---------------------------------------------------------------------------
// Ridge-regression pieces
// ---------------------------------------------------------------------------
__global__ __launch_bounds__(256) void k_colsq(
    const float* __restrict__ x, long xB, float* __restrict__ sq, long sqB,
    int C, int P, long ld) {
  long b = blockIdx.z;
  int p = blockIdx.x * blockDim.x + threadIdx.x;
  if (p < P) {
    const float* xb = x + b * xB;
    float s = 0.0f;
    for (int c = 0; c < C; ++c) { float v = xb[(long)c * ld + p]; s += v * v; }
    sq[b * sqB + p] = s;
  }
}

__global__ __launch_bounds__(256) void k_rbf(
    const float* __restrict__ G, const float* __restrict__ sq,
    const float* __restrict__ Wf, const float* __restrict__ betaP,
    const float* __restrict__ lamP, float* __restrict__ Km,
    float* __restrict__ Am, int Mn) {
  long b = blockIdx.z;
  const float* Gb  = G  + b * (long)Mn * Mn;
  const float* sqb = sq + b * Mn;
  const float* wfb = Wf + b * Mn;
  float* Kb = Km + b * (long)Mn * Mn;
  float* Ab = Am + b * (long)Mn * Mn;
  float beta = betaP[0], lam = lamP[0];
  long tot = (long)Mn * Mn;
  GSTRIDE(i, tot) {
    int m = (int)(i / Mn), k = (int)(i % Mn);
    float d2 = fmaxf(sqb[m] + sqb[k] - 2.0f * Gb[i], 0.0f);
    float kv = __expf(-beta * d2);
    Kb[i] = kv;
    float a = wfb[m] * wfb[k] * kv;
    if (m == k) a += lam;
    Ab[i] = a;
  }
}

__global__ __launch_bounds__(256) void k_rhs(
    const float* __restrict__ eqF, const float* __restrict__ Wf,
    float* __restrict__ rhs, int Mn, int C) {
  long b = blockIdx.z;
  const float* e = eqF + b * (long)C * Mn;
  const float* w = Wf + b * Mn;
  float* r = rhs + b * (long)Mn * C;
  long tot = (long)Mn * C;
  GSTRIDE(i, tot) {
    int m = (int)(i / C), c = (int)(i % C);
    r[i] = w[m] * e[(long)c * Mn + m];
  }
}

// Gauss-Jordan: one block per batch; A (Mn x Mn), RHS (Mn x C) solved in place
__global__ __launch_bounds__(256) void k_solve(
    float* __restrict__ Amat, float* __restrict__ R, int Mn, int C) {
  long b = blockIdx.x;
  float* A  = Amat + b * (long)Mn * Mn;
  float* Rb = R    + b * (long)Mn * C;
  __shared__ float prow[256], prowR[128], fac[256];
  __shared__ float pinv;
  for (int p = 0; p < Mn; ++p) {
    if (threadIdx.x == 0) pinv = 1.0f / A[(long)p * Mn + p];
    __syncthreads();
    for (int j = threadIdx.x; j < Mn; j += blockDim.x) {
      float v = A[(long)p * Mn + j] * pinv; A[(long)p * Mn + j] = v; prow[j] = v;
    }
    for (int j = threadIdx.x; j < C; j += blockDim.x) {
      float v = Rb[(long)p * C + j] * pinv; Rb[(long)p * C + j] = v; prowR[j] = v;
    }
    for (int r = threadIdx.x; r < Mn; r += blockDim.x)
      fac[r] = (r == p) ? 0.0f : A[(long)r * Mn + p];
    __syncthreads();
    for (long t = threadIdx.x; t < (long)Mn * Mn; t += blockDim.x) {
      int r = (int)(t / Mn), j = (int)(t % Mn);
      A[t] -= fac[r] * prow[j];
    }
    for (long t = threadIdx.x; t < (long)Mn * C; t += blockDim.x) {
      int r = (int)(t / C), j = (int)(t % C);
      Rb[t] -= fac[r] * prowR[j];
    }
    __syncthreads();
  }
}

// ===========================================================================
// Host orchestration
// ===========================================================================
struct ConvP { const float *b, *w; };
struct BnP   { const float *b, *g; };
struct AttnP { ConvP cat1, cat2; BnP cat_bn; ConvP k, mh, q, v; };
struct PcnP  { BnP bn1, bn2; ConvP c1, c2; };
struct DmfcP { const float* beta; BnP fw_bn; ConvP fw_conv; AttnP inject;
               const float* lamda; ConvP pos_f; AttnP rect, rect1;
               BnP samp_bn; ConvP samp_conv; ConvP val_f; };
struct LccP  { ConvP down, f1, f2; BnP f_bn; ConvP l1, l2; BnP l_bn;
               ConvP s1, s2; BnP s_bn; ConvP up; };
struct AttnW { half_t *q, *k, *v, *mh, *c1, *c2; };
struct DmfcW { AttnW inject, rect, rect1; half_t *samp, *posf, *val; };

extern "C" void kernel_launch(void* const* d_in, const int* in_sizes, int n_in,
                              void* d_out, int out_size, void* d_ws, size_t ws_size,
                              hipStream_t stream) {
  (void)in_sizes; (void)n_in; (void)out_size; (void)ws_size;
  const int Bn = 8, Cc = 128, Nn = 4000, Mm = 250, KDc = 9, C2 = 64;

  // ---- parse pytree (alphabetical dict order) ----
  int ii = 0;
  auto nextF = [&]() { return (const float*)d_in[ii++]; };
  const float* corr = nextF();
  const int*   idxs = (const int*)d_in[ii++];
  auto conv = [&]() { ConvP c; c.b = nextF(); c.w = nextF(); return c; };
  auto bnp  = [&]() { BnP p; p.b = nextF(); p.g = nextF(); return p; };
  auto attp = [&]() { AttnP a; a.cat1 = conv(); a.cat2 = conv(); a.cat_bn = bnp();
                      a.k = conv(); a.mh = conv(); a.q = conv(); a.v = conv(); return a; };
  auto pcnp = [&]() { PcnP p; p.bn1 = bnp(); p.bn2 = bnp(); p.c1 = conv(); p.c2 = conv(); return p; };
  auto dmp  = [&]() { DmfcP d; d.beta = nextF(); d.fw_bn = bnp(); d.fw_conv = conv();
                      d.inject = attp(); d.lamda = nextF(); d.pos_f = conv();
                      d.rect = attp(); d.rect1 = attp(); d.samp_bn = bnp();
                      d.samp_conv = conv(); d.val_f = conv(); return d; };
  auto lcp  = [&]() { LccP l; l.down = conv(); l.f1 = conv(); l.f2 = conv(); l.f_bn = bnp();
                      l.l1 = conv(); l.l2 = conv(); l.l_bn = bnp();
                      l.s1 = conv(); l.s2 = conv(); l.s_bn = bnp(); l.up = conv(); return l; };
  PcnP cn1 = pcnp(), cn2 = pcnp();
  DmfcP dm1 = dmp(), dm2 = dmp();
  LccP lc1 = lcp(), lc2 = lcp();
  BnP pp_bn = bnp();
  ConvP pp_conv = conv();
  const float* pos = nextF();

  // ---- workspace arena ----
  size_t off = 0;
  auto alloc  = [&](size_t bytes) { void* p = (char*)d_ws + off;
                                    off += (bytes + 255) & ~(size_t)255; return p; };
  auto allocF = [&](long n) { return (float*)alloc((size_t)n * 4); };
  auto allocH = [&](long n) { return (half_t*)alloc((size_t)n * 2); };
  auto gridFor = [&](long n) { long g = (n + 255) / 256; if (g > 4096) g = 4096;
                               return (unsigned)g; };

  // ---- f16 weight conversion ----
  auto cvtW = [&](const float* w, int rows, int cols) {
    half_t* dst = allocH((long)rows * cols);
    k_cvt16<<<dim3(gridFor((long)rows * cols), 1, 1), 256, 0, stream>>>(
        w, 0, dst, 0, rows, cols, cols, cols, 0);
    return dst;
  };
  auto cvtAttnW = [&](AttnP& a) {
    AttnW w; w.q = cvtW(a.q.w, 128, 128); w.k = cvtW(a.k.w, 128, 128);
    w.v = cvtW(a.v.w, 128, 128); w.mh = cvtW(a.mh.w, 128, 128);
    w.c1 = cvtW(a.cat1.w, 256, 256); w.c2 = cvtW(a.cat2.w, 128, 256); return w;
  };
  auto cvtDmW = [&](DmfcP& d) {
    DmfcW w; w.inject = cvtAttnW(d.inject); w.rect = cvtAttnW(d.rect);
    w.rect1 = cvtAttnW(d.rect1); w.samp = cvtW(d.samp_conv.w, Mm, 128);
    w.posf = cvtW(d.pos_f.w, 64, 64); w.val = cvtW(d.val_f.w, 128, 128); return w;
  };
  half_t* c1w1 = cvtW(cn1.c1.w, 128, 128); half_t* c2w1 = cvtW(cn1.c2.w, 128, 128);
  half_t* c1w2 = cvtW(cn2.c1.w, 128, 128); half_t* c2w2 = cvtW(cn2.c2.w, 128, 128);
  DmfcW dmW1 = cvtDmW(dm1), dmW2 = cvtDmW(dm2);

  // ---- activation buffers ----
  const long CN = (long)Cc * Nn;
  float* cf   = allocF((long)Bn * CN);
  float* cf2  = allocF((long)Bn * CN);
  float* t0   = allocF((long)Bn * CN);
  float* t1   = allocF((long)Bn * CN);
  float* lccO = allocF((long)Bn * CN);
  float* f9   = allocF((long)Bn * KDc * Nn);
  float* d0   = allocF((long)Bn * KDc * Nn * KDc);
  float* d1   = allocF((long)Bn * KDc * Nn * KDc);
  float* d2b  = allocF((long)Bn * KDc * Nn * KDc);
  float* qb   = allocF((long)Bn * CN);
  float* kb   = allocF((long)Bn * CN);
  float* vb   = allocF((long)Bn * CN);
  float* addb = allocF((long)Bn * CN);
  float* catb = allocF((long)Bn * 2 * CN);   // also aliases the W-softmax buffer
  float* cat1b= allocF((long)Bn * 2 * CN);
  float* Sb   = allocF((long)Bn * Nn * Mm);
  half_t* qT16= allocH((long)Bn * Nn * Cc);
  half_t* v16b= allocH((long)Bn * CN);
  half_t* a16 = allocH((long)Bn * CN);
  half_t* p16 = allocH((long)Bn * C2 * Nn);
  half_t* x16 = allocH((long)Bn * Mm * C2);
  half_t* K16 = allocH((long)Bn * Mm * 256);
  float* FM   = allocF((long)Bn * Cc * Mm);
  float* PM   = allocF((long)Bn * C2 * Mm);
  float* FR   = allocF((long)Bn * Cc * Mm);
  float* t250 = allocF((long)Bn * Cc * Mm);
  float* pre2 = allocF((long)Bn * Cc * Mm);
  float* preMc= allocF((long)Bn * Mm * Cc);
  float* Wfb  = allocF((long)Bn * Mm);
  float* x64  = allocF((long)Bn * C2 * Mm);
  float* sqv  = allocF((long)Bn * Mm);
  float* Gm   = allocF((long)Bn * Mm * Mm);
  float* Km   = allocF((long)Bn * Mm * Mm);
  float* Am   = allocF((long)Bn * Mm * Mm);
  float* eqF  = allocF((long)Bn * Cc * Mm);
  float* rhs  = allocF((long)Bn * Mm * Cc);
  float* scl  = allocF(2048);
  float* shf  = allocF(2048);

  // ---- launch helpers ----
  auto gemmW = [&](const half_t* A, long lda, long aB,
                   const float* Bp, long bsK, long bsN, long bB,
                   const float* bias, const float* resid, long resB,
                   float* Yp, long ldy, long yB, int M, int K, int P,
                   float alpha, int relu) {
    dim3 g((P + 15) / 16, (M + 127) / 128, Bn);
    k_wmma_gemm<<<g, 256, 0, stream>>>(A, lda, aB, Bp, bsK, bsN, bB, bias,
                                       resid, resB, Yp, ldy, yB, M, K, P, alpha, relu);
  };
  auto gemmS = [&](const float* W, const float* bias, const float* X, long xB, long ldx,
                   const float* resid, long resB, float* Yp, long yB, long ldy,
                   int M, int K, int P, int act) {
    k_gemm_small<<<dim3(gridFor((long)M * P), 1, Bn), 256, 0, stream>>>(
        W, bias, X, xB, ldx, resid, resB, Yp, yB, ldy, M, K, P, act);
  };
  auto cvtAct = [&](const float* s, long sB, half_t* d, long dB, int rows, int cols,
                    long sLd, long dLd, int tr) {
    k_cvt16<<<dim3(gridFor((long)rows * cols), 1, Bn), 256, 0, stream>>>(
        s, sB, d, dB, rows, cols, sLd, dLd, tr);
  };
  auto statsAffine = [&](const float* x, float* y, int C, long L,
                         const BnP* p, float eps, int relu) {
    k_stats<<<dim3(C, 1, 1), 256, 0, stream>>>(x, L, (long)C * L, Bn, L,
                                               p ? p->g : nullptr, p ? p->b : nullptr,
                                               eps, scl, shf);
    long tot = (long)Bn * C * L;
    k_affine<<<dim3(gridFor(tot)), 256, 0, stream>>>(x, y, scl, shf, tot, L, C, relu);
  };
  auto inorm = [&](const float* x, float* y, int C, long L) {
    k_stats<<<dim3(Bn * C, 1, 1), 256, 0, stream>>>(x, L, 0, 1, L, nullptr, nullptr,
                                                    1e-3f, scl, shf);
    long tot = (long)Bn * C * L;
    k_affine<<<dim3(gridFor(tot)), 256, 0, stream>>>(x, y, scl, shf, tot, L, Bn * C, 0);
  };

  // ---- attention block ----
  auto attention = [&](AttnP& P_, AttnW& Wt, const float* m1, long m1B, int n1,
                       const float* m2, long m2B, int n2, float* out, long outB) {
    gemmW(Wt.q, 128, 0, m1, n1, 1, m1B, P_.q.b, nullptr, 0, qb, n1, (long)128 * n1,
          128, 128, n1, 1.0f, 0);
    gemmW(Wt.k, 128, 0, m2, n2, 1, m2B, P_.k.b, nullptr, 0, kb, n2, (long)128 * n2,
          128, 128, n2, 1.0f, 0);
    gemmW(Wt.v, 128, 0, m2, n2, 1, m2B, P_.v.b, nullptr, 0, vb, n2, (long)128 * n2,
          128, 128, n2, 1.0f, 0);
    cvtAct(qb, (long)128 * n1, qT16, (long)n1 * 128, 128, n1, n1, 128, 1);
    int vld = (n2 + 7) & ~7;
    cvtAct(vb, (long)128 * n2, v16b, (long)128 * vld, 128, n2, n2, vld, 0);
    const float sc = 0.1767766953f;  // 1/sqrt(HD=32)
    for (int h = 0; h < 4; ++h) {
      gemmW(qT16 + h * 32, 128, (long)n1 * 128,
            kb + (long)h * 32 * n2, n2, 1, (long)128 * n2,
            nullptr, nullptr, 0, Sb, n2, (long)n1 * n2, n1, 32, n2, sc, 0);
      k_softmax<<<dim3((unsigned)(Bn * n1)), 256, 0, stream>>>(Sb, n2, n2);
      gemmW(v16b + (long)h * 32 * vld, vld, (long)128 * vld,
            Sb, 1, n2, (long)n1 * n2, nullptr, nullptr, 0,
            addb + (long)h * 32 * n1, n1, (long)128 * n1, 32, n2, n1, 1.0f, 0);
    }
    gemmW(Wt.mh, 128, 0, addb, n1, 1, (long)128 * n1, P_.mh.b, nullptr, 0,
          catb + (long)128 * n1, n1, (long)256 * n1, 128, 128, n1, 1.0f, 0);
    k_copy<<<dim3(gridFor((long)128 * n1), 1, Bn), 256, 0, stream>>>(
        m1, m1B, catb, (long)256 * n1, (long)128 * n1);
    gemmW(Wt.c1, 256, 0, catb, n1, 1, (long)256 * n1, P_.cat1.b, nullptr, 0,
          cat1b, n1, (long)256 * n1, 256, 256, n1, 1.0f, 0);
    k_stats<<<dim3(256, 1, 1), 256, 0, stream>>>(cat1b, n1, (long)256 * n1, Bn, n1,
                                                 P_.cat_bn.g, P_.cat_bn.b, 1e-5f, scl, shf);
    { long tot = (long)Bn * 256 * n1;
      k_affine<<<dim3(gridFor(tot)), 256, 0, stream>>>(cat1b, cat1b, scl, shf, tot, n1, 256, 1); }
    gemmW(Wt.c2, 256, 0, cat1b, n1, 1, (long)256 * n1, P_.cat2.b, m1, m1B,
          out, n1, outB, 128, 256, n1, 1.0f, 0);
  };

  // ---- LCC block (feats + lcc(feats) -> outAdd) ----
  auto lccBlock = [&](LccP& L_, const float* feats, float* outAdd) {
    const long L36 = (long)Nn * KDc;
    gemmS(L_.down.w, L_.down.b, feats, CN, Nn, nullptr, 0, f9, (long)KDc * Nn, Nn,
          KDc, Cc, Nn, 0);
    k_gather_diff<<<dim3(gridFor((long)Bn * KDc * Nn * KDc)), 256, 0, stream>>>(
        f9, idxs, d0, Bn, KDc, Nn, KDc);
    gemmS(L_.f1.w, L_.f1.b, d0, KDc * L36, L36, nullptr, 0, d1, KDc * L36, L36,
          KDc, KDc, (int)L36, 0);
    statsAffine(d1, d1, KDc, L36, &L_.f_bn, 1e-5f, 1);
    gemmS(L_.f2.w, L_.f2.b, d1, KDc * L36, L36, d0, KDc * L36, d2b, KDc * L36, L36,
          KDc, KDc, (int)L36, 0);
    k_perm<<<dim3(gridFor((long)Bn * KDc * Nn * KDc)), 256, 0, stream>>>(
        d2b, d0, Bn, KDc, Nn, KDc);
    gemmS(L_.s1.w, L_.s1.b, d0, KDc * L36, L36, nullptr, 0, d1, KDc * L36, L36,
          KDc, KDc, (int)L36, 0);
    statsAffine(d1, d1, KDc, L36, &L_.s_bn, 1e-5f, 1);
    gemmS(L_.s2.w, L_.s2.b, d1, KDc * L36, L36, d0, KDc * L36, d2b, KDc * L36, L36,
          KDc, KDc, (int)L36, 0);
    gemmS(L_.l1.w, L_.l1.b, d2b, KDc * L36, L36, nullptr, 0, d1, KDc * L36, L36,
          KDc, KDc, (int)L36, 0);
    statsAffine(d1, d1, KDc, L36, &L_.l_bn, 1e-5f, 1);
    gemmS(L_.l2.w, L_.l2.b, d1, KDc * L36, L36, nullptr, 0, f9, L36, L36,
          1, KDc, (int)L36, 0);
    k_tr2<<<dim3(gridFor((long)Bn * Nn * KDc)), 256, 0, stream>>>(f9, d1, Bn, Nn, KDc);
    gemmS(L_.up.w, L_.up.b, d1, (long)KDc * Nn, Nn, nullptr, 0, lccO, CN, Nn,
          Cc, KDc, Nn, 0);
    k_axpy<<<dim3(gridFor((long)Bn * CN)), 256, 0, stream>>>(feats, lccO, outAdd,
                                                             (long)Bn * CN);
  };

  // ---- PointCN block (in-place on x) ----
  auto pointcn = [&](PcnP& P_, half_t* c1w, half_t* c2w, float* x) {
    inorm(x, t0, Cc, Nn);
    statsAffine(t0, t1, Cc, Nn, &P_.bn1, 1e-5f, 1);
    gemmW(c1w, 128, 0, t1, Nn, 1, CN, P_.c1.b, nullptr, 0, t0, Nn, CN,
          128, 128, Nn, 1.0f, 0);
    inorm(t0, t0, Cc, Nn);
    statsAffine(t0, t1, Cc, Nn, &P_.bn2, 1e-5f, 1);
    gemmW(c2w, 128, 0, t1, Nn, 1, CN, P_.c2.b, x, CN, x, Nn, CN,
          128, 128, Nn, 1.0f, 0);
  };

  // ---- DMFC block ----
  auto dmfcBlock = [&](DmfcP& D_, DmfcW& W_, float* cfio, float* outCf) {
    inorm(cfio, t0, Cc, Nn);
    statsAffine(t0, t1, Cc, Nn, &D_.samp_bn, 1e-5f, 1);
    float* wb = catb;  // alias: consumed before attention reuses catb
    gemmW(W_.samp, 128, 0, t1, Nn, 1, CN, D_.samp_conv.b, nullptr, 0,
          wb, Nn, (long)Mm * Nn, Mm, Cc, Nn, 1.0f, 0);
    k_softmax<<<dim3((unsigned)(Bn * Mm)), 256, 0, stream>>>(wb, Nn, Nn);
    cvtAct(cfio, CN, a16, CN, Cc, Nn, Nn, Nn, 0);
    gemmW(a16, Nn, CN, wb, 1, Nn, (long)Mm * Nn, nullptr, nullptr, 0,
          FM, Mm, (long)Cc * Mm, Cc, Nn, Mm, 1.0f, 0);
    cvtAct(pos, (long)C2 * Nn, p16, (long)C2 * Nn, C2, Nn, Nn, Nn, 0);
    gemmW(p16, Nn, (long)C2 * Nn, wb, 1, Nn, (long)Mm * Nn, nullptr, nullptr, 0,
          PM, Mm, (long)C2 * Mm, C2, Nn, Mm, 1.0f, 0);
    attention(D_.inject, W_.inject, FM, (long)Cc * Mm, Mm, cfio, CN, Nn,
              FR, (long)Cc * Mm);
    statsAffine(FR, t250, Cc, Mm, &D_.fw_bn, 1e-5f, 1);
    gemmS(D_.fw_conv.w, D_.fw_conv.b, t250, (long)Cc * Mm, Mm, nullptr, 0,
          Wfb, Mm, Mm, 1, Cc, Mm, 2);
    gemmW(W_.posf, C2, 0, PM, Mm, 1, (long)C2 * Mm, D_.pos_f.b, nullptr, 0,
          x64, Mm, (long)C2 * Mm, C2, C2, Mm, 1.0f, 0);
    k_colsq<<<dim3(1, 1, Bn), 256, 0, stream>>>(x64, (long)C2 * Mm, sqv, Mm,
                                                C2, Mm, Mm);
    cvtAct(x64, (long)C2 * Mm, x16, (long)Mm * C2, C2, Mm, Mm, C2, 1);
    gemmW(x16, C2, (long)Mm * C2, x64, Mm, 1, (long)C2 * Mm, nullptr, nullptr, 0,
          Gm, Mm, (long)Mm * Mm, Mm, C2, Mm, 1.0f, 0);
    k_rbf<<<dim3(gridFor((long)Mm * Mm), 1, Bn), 256, 0, stream>>>(
        Gm, sqv, Wfb, D_.beta, D_.lamda, Km, Am, Mm);
    gemmW(W_.val, 128, 0, FR, Mm, 1, (long)Cc * Mm, D_.val_f.b, nullptr, 0,
          eqF, Mm, (long)Cc * Mm, Cc, Cc, Mm, 1.0f, 0);
    k_rhs<<<dim3(gridFor((long)Mm * Cc), 1, Bn), 256, 0, stream>>>(eqF, Wfb, rhs, Mm, Cc);
    k_solve<<<dim3(Bn), 256, 0, stream>>>(Am, rhs, Mm, Cc);
    cvtAct(Km, (long)Mm * Mm, K16, (long)Mm * 256, Mm, Mm, Mm, 256, 0);
    gemmW(K16, 256, (long)Mm * 256, rhs, Cc, 1, (long)Mm * Cc, nullptr, nullptr, 0,
          preMc, Cc, (long)Mm * Cc, Mm, Mm, Cc, 1.0f, 0);
    k_tr32<<<dim3(gridFor((long)Mm * Cc), 1, Bn), 256, 0, stream>>>(
        preMc, (long)Mm * Cc, t250, (long)Cc * Mm, Mm, Cc);
    attention(D_.rect1, W_.rect1, FR, (long)Cc * Mm, Mm, t250, (long)Cc * Mm, Mm,
              pre2, (long)Cc * Mm);
    attention(D_.rect, W_.rect, cfio, CN, Nn, pre2, (long)Cc * Mm, Mm,
              outCf, CN);
  };

  // ---- full pipeline ----
  lccBlock(lc1, corr, cf);
  pointcn(cn1, c1w1, c2w1, cf);
  dmfcBlock(dm1, dmW1, cf, cf2);
  { float* t = cf; cf = cf2; cf2 = t; }
  lccBlock(lc2, cf, cf);
  pointcn(cn2, c1w2, c2w2, cf);
  dmfcBlock(dm2, dmW2, cf, cf2);
  { float* t = cf; cf = cf2; cf2 = t; }
  statsAffine(cf, t0, Cc, Nn, &pp_bn, 1e-5f, 1);
  gemmS(pp_conv.w, pp_conv.b, t0, CN, Nn, nullptr, 0, (float*)d_out, Nn, Nn,
        1, Cc, Nn, 0);
}
